// ViT_Autoencoder_69870527972102
// MI455X (gfx1250) — compile-verified, matrix operand order checked by reading, not run on hardware
//
#include <hip/hip_runtime.h>
#include <hip/hip_bf16.h>
#include <math.h>

// ===========================================================================
// ViT autoencoder forward for gfx1250 (MI455X), bf16 WMMA everywhere.
//
// Input leaf order assumption: JAX pytree flattening (dicts in sorted key
// order):
//   0: images (32,3,224,224)
//   1: params.cls (1,768)
//   2..49:  params.dec[i] leaves, 12 per block, sorted:
//           [b1,b2,in_b,in_w,ln1_b,ln1_g,ln2_b,ln2_g,out_b,out_w,w1,w2]
//   50: dec_map_b  51: dec_map_w  52: dec_pos
//   53..148: params.enc[i] leaves (same 12-leaf order), 8 blocks
//   149: map_b  150: map_w  151: pos_embed (197,768)
// Workspace requirement: 6304*(768*3 + 3072)*4 B ~= 136 MB.
// ===========================================================================

typedef __bf16 bf16;
typedef float v8f   __attribute__((ext_vector_type(8)));
typedef bf16  v16bf __attribute__((ext_vector_type(16)));
typedef bf16  v8bf  __attribute__((ext_vector_type(8)));

union Frag { v16bf v; v8bf h[2]; };

__device__ __forceinline__ v8f wmma_bf16(const Frag& a, const Frag& b, v8f c) {
  // D = A(16x32) * B(32x16) + C, fp32 accumulate
  return __builtin_amdgcn_wmma_f32_16x16x32_bf16(false, a.v, false, b.v,
                                                 (short)0, c, false, false);
}

#define TILE_M 128
#define TILE_N 64
#define TILE_K 64
#define LDSW   72  // 64 + 8 bf16 pad -> 144B rows, conflict-free b128 frag loads
#define NLA    8   // A-tile float4 loads per thread (128*64/4/256)
#define NLB    4   // B-tile float4 loads per thread ( 64*64/4/256)

// C[M,N] = A[M,K] * W[N,K]^T (+bias[n]) (->gelu) (+res[(r%mod or r), n])
// SKIP: A row = r + r/196 + 1 (drops cls rows: b*196+s -> b*197+s+1)
// GUARD: M not a multiple of TILE_M (only the qkv/enc GEMMs with M=6304)
template<bool GUARD, bool SKIP>
__global__ __launch_bounds__(256)
void gemm_bf16_kernel(const float* __restrict__ A, const float* __restrict__ Wt,
                      const float* __restrict__ bias, const float* __restrict__ res,
                      float* __restrict__ out,
                      int M, int N, int K, int res_mod, int gelu_flag)
{
  __shared__ bf16 sA[2][TILE_M * LDSW];
  __shared__ bf16 sB[2][TILE_N * LDSW];

  const int tid  = threadIdx.x;
  const int bn   = blockIdx.x;
  const int bm   = blockIdx.y;
  const int wave = tid >> 5;
  const int lane = tid & 31;
  const int hh   = lane >> 4;   // lane half (0/1)
  const int l16  = lane & 15;
  const int wm   = wave >> 1;   // 0..3 (M)
  const int wn   = wave & 1;    // 0..1 (N)

  // ---- loop-invariant staging descriptors (pointers + LDS offsets) --------
  const float* aptr[NLA]; int aoff[NLA]; bool aval[NLA];
#pragma unroll
  for (int i = 0; i < NLA; ++i) {
    int id = tid + i * 256;
    int r  = id >> 4;           // 16 float4 per 64-wide row
    int c4 = id & 15;
    int gr = bm * TILE_M + r;
    aval[i] = !GUARD || (gr < M);
    int ar  = SKIP ? (gr + gr / 196 + 1) : gr;
    aptr[i] = A + (size_t)ar * K + c4 * 4;
    aoff[i] = r * LDSW + c4 * 4;
  }
  const float* bptr[NLB]; int boff[NLB];
#pragma unroll
  for (int i = 0; i < NLB; ++i) {
    int id = tid + i * 256;
    int r  = id >> 4;
    int c4 = id & 15;
    bptr[i] = Wt + (size_t)(bn * TILE_N + r) * K + c4 * 4;
    boff[i] = r * LDSW + c4 * 4;
  }

  const v8f vzero = {0.f,0.f,0.f,0.f,0.f,0.f,0.f,0.f};
  v8f acc[2][2] = {{vzero, vzero}, {vzero, vzero}};

  // ---- stage tile 0 -------------------------------------------------------
#pragma unroll
  for (int i = 0; i < NLA; ++i) {
    float4 x = make_float4(0.f,0.f,0.f,0.f);
    if (!GUARD || aval[i]) x = *(const float4*)(aptr[i]);
    bf16* d = &sA[0][aoff[i]];
    d[0]=(bf16)x.x; d[1]=(bf16)x.y; d[2]=(bf16)x.z; d[3]=(bf16)x.w;
  }
#pragma unroll
  for (int i = 0; i < NLB; ++i) {
    float4 x = *(const float4*)(bptr[i]);
    bf16* d = &sB[0][boff[i]];
    d[0]=(bf16)x.x; d[1]=(bf16)x.y; d[2]=(bf16)x.z; d[3]=(bf16)x.w;
  }
  __syncthreads();

  const int T = K / TILE_K;
  for (int t = 0; t < T; ++t) {
    // ---- fetch tile t+1 into registers (overlaps with WMMA below) --------
    float4 ra[NLA], rb[NLB];
    const bool more = (t + 1 < T);
    if (more) {
      const int k1 = (t + 1) * TILE_K;
#pragma unroll
      for (int i = 0; i < NLA; ++i) {
        ra[i] = make_float4(0.f,0.f,0.f,0.f);
        if (!GUARD || aval[i]) ra[i] = *(const float4*)(aptr[i] + k1);
      }
#pragma unroll
      for (int i = 0; i < NLB; ++i) rb[i] = *(const float4*)(bptr[i] + k1);
      if (t + 2 < T) {  // warm L2/WGP$ two tiles ahead (global_prefetch_b8)
        __builtin_prefetch(aptr[0] + (t + 2) * TILE_K, 0, 3);
        __builtin_prefetch(bptr[0] + (t + 2) * TILE_K, 0, 3);
      }
    }

    // ---- compute from LDS buffer t&1 -------------------------------------
    const bf16* pA0 = &sA[t & 1][(wm*32 + l16) * LDSW];
    const bf16* pA1 = pA0 + 16*LDSW;
    const bf16* pB0 = &sB[t & 1][(wn*32 + l16) * LDSW];
    const bf16* pB1 = pB0 + 16*LDSW;
#pragma unroll
    for (int kk = 0; kk < TILE_K; kk += 32) {
      const int o0 = kk + 8*hh;        // K layout: (j&7) + 8*half + 16*(j>>3)
      const int o1 = kk + 16 + 8*hh;
      Frag a0,a1,b0,b1;
      a0.h[0] = *(const v8bf*)(pA0 + o0); a0.h[1] = *(const v8bf*)(pA0 + o1);
      a1.h[0] = *(const v8bf*)(pA1 + o0); a1.h[1] = *(const v8bf*)(pA1 + o1);
      b0.h[0] = *(const v8bf*)(pB0 + o0); b0.h[1] = *(const v8bf*)(pB0 + o1);
      b1.h[0] = *(const v8bf*)(pB1 + o0); b1.h[1] = *(const v8bf*)(pB1 + o1);
      acc[0][0] = wmma_bf16(a0,b0,acc[0][0]);
      acc[0][1] = wmma_bf16(a0,b1,acc[0][1]);
      acc[1][0] = wmma_bf16(a1,b0,acc[1][0]);
      acc[1][1] = wmma_bf16(a1,b1,acc[1][1]);
    }

    // ---- stage tile t+1 into the other LDS buffer ------------------------
    if (more) {
      const int nb = (t + 1) & 1;
#pragma unroll
      for (int i = 0; i < NLA; ++i) {
        bf16* d = &sA[nb][aoff[i]];
        d[0]=(bf16)ra[i].x; d[1]=(bf16)ra[i].y; d[2]=(bf16)ra[i].z; d[3]=(bf16)ra[i].w;
      }
#pragma unroll
      for (int i = 0; i < NLB; ++i) {
        bf16* d = &sB[nb][boff[i]];
        d[0]=(bf16)rb[i].x; d[1]=(bf16)rb[i].y; d[2]=(bf16)rb[i].z; d[3]=(bf16)rb[i].w;
      }
    }
    __syncthreads();
  }

  // ---- epilogue: C/D layout -> lane=col, vgpr v -> row v + 8*half ---------
#pragma unroll
  for (int tm = 0; tm < 2; ++tm)
#pragma unroll
  for (int tn = 0; tn < 2; ++tn)
#pragma unroll
  for (int v = 0; v < 8; ++v) {
    int r = bm*TILE_M + wm*32 + tm*16 + v + 8*hh;
    int c = bn*TILE_N + wn*32 + tn*16 + l16;
    if (!GUARD || (r < M)) {
      float x = acc[tm][tn][v];
      if (bias) x += bias[c];
      if (gelu_flag) x = 0.5f * x * (1.f + erff(x * 0.70710678118f));
      if (res) {
        int rr = (res_mod > 0) ? (r % res_mod) : r;
        x += res[(size_t)rr * N + c];
      }
      out[(size_t)r * N + c] = x;
    }
  }
}

// Batch-attention: per (s, head) a 32x32 score matrix over the batch dim.
// scores[b,c] = sum_d q[b,s,h,d]*k[c,s,h,d]*scale ; softmax over c ;
// o[b,d] = sum_c attn[b,c]*v[c,d].  One block (4 waves) per (s,h), all in LDS.
template<int DH>
__global__ __launch_bounds__(128)
void attn_kernel(const float* __restrict__ qkv, float* __restrict__ o,
                 int stok, int dmodel)
{
  constexpr int QKS = DH + 8;           // padded bf16 row stride
  __shared__ bf16  sq [32 * QKS];       // q (scaled) [b][d]
  __shared__ bf16  sk [32 * QKS];       // k [c][d]
  __shared__ bf16  svT[DH * 40];        // v^T [d][c]
  __shared__ float ss [32 * 33];        // scores f32 [b][c]
  __shared__ bf16  sp [32 * 40];        // softmax probs bf16 [b][c]

  const int tid = threadIdx.x;
  const int s   = blockIdx.x / 12;
  const int h   = blockIdx.x % 12;
  const int d3  = 3 * dmodel;
  const float qscale = rsqrtf((float)DH);

  for (int id = tid; id < 32*DH/4; id += 128) {
    int r  = id / (DH/4);
    int c4 = id % (DH/4);
    size_t base = ((size_t)r * stok + s) * d3 + h*DH + c4*4;
    float4 q4 = *(const float4*)(qkv + base);
    float4 k4 = *(const float4*)(qkv + base + dmodel);
    float4 v4 = *(const float4*)(qkv + base + 2*dmodel);
    bf16* dq = &sq[r*QKS + c4*4];
    dq[0]=(bf16)(q4.x*qscale); dq[1]=(bf16)(q4.y*qscale);
    dq[2]=(bf16)(q4.z*qscale); dq[3]=(bf16)(q4.w*qscale);
    bf16* dk = &sk[r*QKS + c4*4];
    dk[0]=(bf16)k4.x; dk[1]=(bf16)k4.y; dk[2]=(bf16)k4.z; dk[3]=(bf16)k4.w;
    svT[(c4*4+0)*40 + r] = (bf16)v4.x;
    svT[(c4*4+1)*40 + r] = (bf16)v4.y;
    svT[(c4*4+2)*40 + r] = (bf16)v4.z;
    svT[(c4*4+3)*40 + r] = (bf16)v4.w;
  }
  __syncthreads();

  const int wave = tid >> 5, lane = tid & 31, hh = lane >> 4, l16 = lane & 15;
  const v8f vzero = {0.f,0.f,0.f,0.f,0.f,0.f,0.f,0.f};

  { // scores = q * k^T : 2x2 tiles of 16x16, one tile per wave
    const int mt = wave >> 1, nt = wave & 1;
    v8f acc = vzero;
    const bf16* pq = &sq[(mt*16 + l16) * QKS];
    const bf16* pk = &sk[(nt*16 + l16) * QKS];
#pragma unroll
    for (int kk = 0; kk < DH; kk += 32) {
      Frag a, b;
      a.h[0]=*(const v8bf*)(pq+kk+8*hh); a.h[1]=*(const v8bf*)(pq+kk+16+8*hh);
      b.h[0]=*(const v8bf*)(pk+kk+8*hh); b.h[1]=*(const v8bf*)(pk+kk+16+8*hh);
      acc = wmma_bf16(a, b, acc);
    }
#pragma unroll
    for (int v = 0; v < 8; ++v)
      ss[(mt*16 + v + 8*hh)*33 + nt*16 + l16] = acc[v];
  }
  __syncthreads();

  if (tid < 32) { // softmax over c (batch dim, 32 wide)
    float m = -1e30f;
    for (int c = 0; c < 32; ++c) m = fmaxf(m, ss[tid*33+c]);
    float sum = 0.f;
    for (int c = 0; c < 32; ++c) sum += __expf(ss[tid*33+c] - m);
    float inv = 1.f / sum;
    for (int c = 0; c < 32; ++c)
      sp[tid*40 + c] = (bf16)(__expf(ss[tid*33+c] - m) * inv);
  }
  __syncthreads();

  // o = attn(32x32) * v(32xDH) : tiles 2 x (DH/16), round-robin over waves
  constexpr int NT = DH/16;
  for (int t = wave; t < 2*NT; t += 4) {
    int mt = t / NT, nt = t % NT;
    Frag a, b;
    const bf16* pa = &sp [(mt*16 + l16)*40];
    const bf16* pb = &svT[(nt*16 + l16)*40];
    a.h[0]=*(const v8bf*)(pa+8*hh); a.h[1]=*(const v8bf*)(pa+16+8*hh);
    b.h[0]=*(const v8bf*)(pb+8*hh); b.h[1]=*(const v8bf*)(pb+16+8*hh);
    v8f c = vzero;
    c = wmma_bf16(a, b, c);
#pragma unroll
    for (int v = 0; v < 8; ++v) {
      int br = mt*16 + v + 8*hh;
      int d  = nt*16 + l16;
      o[((size_t)br * stok + s) * dmodel + h*DH + d] = c[v];
    }
  }
}

// One wave32 per row; float4 loads; shuffle reduction. d % 128 == 0.
__global__ __launch_bounds__(256)
void ln_kernel(const float* __restrict__ x, float* __restrict__ y,
               const float* __restrict__ g, const float* __restrict__ b,
               int rows, int d)
{
  const int wave = threadIdx.x >> 5, lane = threadIdx.x & 31;
  const int row = blockIdx.x * 8 + wave;
  if (row >= rows) return;
  const float* xr = x + (size_t)row * d;
  float s = 0.f, s2 = 0.f;
  for (int i = lane*4; i < d; i += 128) {
    float4 v = *(const float4*)(xr + i);
    s  += v.x + v.y + v.z + v.w;
    s2 += v.x*v.x + v.y*v.y + v.z*v.z + v.w*v.w;
  }
  for (int m = 16; m > 0; m >>= 1) {
    s  += __shfl_xor(s,  m, 32);
    s2 += __shfl_xor(s2, m, 32);
  }
  float mean = s / d;
  float inv  = rsqrtf(s2 / d - mean*mean + 1e-5f);
  float* yr = y + (size_t)row * d;
  for (int i = lane*4; i < d; i += 128) {
    float4 v  = *(const float4*)(xr + i);
    float4 vg = *(const float4*)(g + i);
    float4 vb = *(const float4*)(b + i);
    float4 o;
    o.x = (v.x - mean) * inv * vg.x + vb.x;
    o.y = (v.y - mean) * inv * vg.y + vb.y;
    o.z = (v.z - mean) * inv * vg.z + vb.z;
    o.w = (v.w - mean) * inv * vg.w + vb.w;
    *(float4*)(yr + i) = o;
  }
}

// images (32,3,224,224) -> patch matrix (32*196, 768) with 16x16 patches
__global__ void patchify_kernel(const float* __restrict__ img, float* __restrict__ P)
{
  int idx = blockIdx.x * 256 + threadIdx.x;
  if (idx >= 32*196*768) return;
  int col = idx % 768;  int rowm = idx / 768;
  int s = rowm % 196;   int b = rowm / 196;
  int c = col / 256;    int rem = col % 256;
  int ph = rem / 16;    int pw = rem % 16;
  int sr = s / 14, sc = s % 14;
  P[idx] = img[((size_t)(b*3 + c)*224 + sr*16 + ph)*224 + sc*16 + pw];
}

// x(32,197,768) = concat(cls, tokens) + pos
__global__ void assemble_kernel(const float* __restrict__ tokens,
                                const float* __restrict__ cls,
                                const float* __restrict__ pos,
                                float* __restrict__ x)
{
  int idx = blockIdx.x * 256 + threadIdx.x;
  if (idx >= 32*197*768) return;
  int col = idx % 768;  int row = idx / 768;
  int t = row % 197;    int b = row / 197;
  float v = (t == 0) ? cls[col]
                     : tokens[((size_t)b*196 + (t-1))*768 + col];
  x[idx] = v + pos[t*768 + col];
}

extern "C" void kernel_launch(void* const* d_in, const int* in_sizes, int n_in,
                              void* d_out, int out_size, void* d_ws, size_t ws_size,
                              hipStream_t stream)
{
  (void)in_sizes; (void)n_in; (void)out_size; (void)ws_size;
  auto F = [&](int i) { return (const float*)d_in[i]; };

  const float* images    = F(0);
  const float* cls       = F(1);
  const float* dec_map_b = F(50);
  const float* dec_map_w = F(51);
  const float* dec_pos   = F(52);
  const float* map_b     = F(149);
  const float* map_w     = F(150);
  const float* pos       = F(151);

  struct Blk { const float *b1,*b2,*in_b,*in_w,*ln1_b,*ln1_g,*ln2_b,*ln2_g,
                           *out_b,*out_w,*w1,*w2; };
  auto blk = [&](int base) {
    Blk p; p.b1=F(base+0); p.b2=F(base+1); p.in_b=F(base+2); p.in_w=F(base+3);
    p.ln1_b=F(base+4); p.ln1_g=F(base+5); p.ln2_b=F(base+6); p.ln2_g=F(base+7);
    p.out_b=F(base+8); p.out_w=F(base+9); p.w1=F(base+10); p.w2=F(base+11);
    return p;
  };

  const int Menc = 32*197;   // 6304
  const int Mdec = 32*196;   // 6272
  float* X   = (float*)d_ws;                       // 6304*768
  float* XN  = X  + (size_t)Menc*768;              // 6304*768
  float* O   = XN + (size_t)Menc*768;              // 6304*768
  float* BIG = O  + (size_t)Menc*768;              // 6304*3072

  auto gemm = [&](const float* A, const float* W, const float* bias,
                  const float* res, float* out, int M, int N, int K,
                  int res_mod, int gelu, int skip) {
    dim3 grid(N/64, (M + TILE_M - 1)/TILE_M);
    if (skip)
      gemm_bf16_kernel<false,true ><<<grid, dim3(256), 0, stream>>>(
          A, W, bias, res, out, M, N, K, res_mod, gelu);
    else if (M % TILE_M)
      gemm_bf16_kernel<true ,false><<<grid, dim3(256), 0, stream>>>(
          A, W, bias, res, out, M, N, K, res_mod, gelu);
    else
      gemm_bf16_kernel<false,false><<<grid, dim3(256), 0, stream>>>(
          A, W, bias, res, out, M, N, K, res_mod, gelu);
  };
  auto ln = [&](const float* x, float* y, const float* g, const float* b,
                int rows, int d) {
    ln_kernel<<<dim3((rows+7)/8), dim3(256), 0, stream>>>(x, y, g, b, rows, d);
  };

  // patchify -> BIG ; tokens = patch @ map_w^T + map_b -> O ; assemble -> X
  patchify_kernel<<<dim3((Mdec*768 + 255)/256), dim3(256), 0, stream>>>(images, BIG);
  gemm(BIG, map_w, map_b, nullptr, O, Mdec, 768, 768, 0, 0, 0);
  assemble_kernel<<<dim3((Menc*768 + 255)/256), dim3(256), 0, stream>>>(O, cls, pos, X);

  // ---- encoder: D=768, dh=64, S=197
  for (int i = 0; i < 8; ++i) {
    Blk p = blk(53 + 12*i);
    ln(X, XN, p.ln1_g, p.ln1_b, Menc, 768);
    gemm(XN, p.in_w, p.in_b, nullptr, BIG, Menc, 2304, 768, 0, 0, 0);     // qkv
    attn_kernel<64><<<dim3(197*12), dim3(128), 0, stream>>>(BIG, O, 197, 768);
    gemm(O, p.out_w, p.out_b, XN, X, Menc, 768, 768, 0, 0, 0);            // out = xn + proj
    ln(X, XN, p.ln2_g, p.ln2_b, Menc, 768);
    gemm(XN, p.w1, p.b1, nullptr, BIG, Menc, 3072, 768, 0, 1, 0);         // gelu(mlp1)
    gemm(BIG, p.w2, p.b2, X, X, Menc, 768, 3072, 0, 0, 0);                // out + mlp2
  }

  // ---- decoder map: drop cls via row-skip, + dec_pos broadcast (mod 196)
  gemm(X, dec_map_w, dec_map_b, dec_pos, O, Mdec, 384, 768, 196, 0, 196);

  // ---- decoder: D=384, dh=32, S=196
  float* dx = O; float* dxn = XN; float* dtmp = X;
  for (int i = 0; i < 4; ++i) {
    Blk p = blk(2 + 12*i);
    ln(dx, dxn, p.ln1_g, p.ln1_b, Mdec, 384);
    gemm(dxn, p.in_w, p.in_b, nullptr, BIG, Mdec, 1152, 384, 0, 0, 0);
    attn_kernel<32><<<dim3(196*12), dim3(128), 0, stream>>>(BIG, dtmp, 196, 384);
    gemm(dtmp, p.out_w, p.out_b, dxn, dx, Mdec, 384, 384, 0, 0, 0);
    ln(dx, dxn, p.ln2_g, p.ln2_b, Mdec, 384);
    gemm(dxn, p.w1, p.b1, nullptr, BIG, Mdec, 1536, 384, 0, 1, 0);
    float* outp = (i == 3) ? (float*)d_out : dx;
    gemm(BIG, p.w2, p.b2, dx, outp, Mdec, 384, 1536, 0, 0, 0);
  }
}
